// Attention_1709396984003
// MI455X (gfx1250) — matrix operand order repaired and checked by reading, not compiled, not run on hardware
//
#include <hip/hip_runtime.h>

#define HEADS 12
#define BATCH 8
#define SEQ   1024
#define CH    768
#define HD    64          // head dim
#define OC    (3*CH)      // qkv output channels

typedef __attribute__((ext_vector_type(16))) __bf16 v16bf;
typedef __attribute__((ext_vector_type(8)))  __bf16 v8bf;
typedef __attribute__((ext_vector_type(4)))  __bf16 v4bf;
typedef __attribute__((ext_vector_type(8)))  float  v8f;
typedef __attribute__((ext_vector_type(4)))  unsigned int v4u;
typedef __attribute__((ext_vector_type(8)))  int    v8i;
typedef __attribute__((ext_vector_type(4)))  int    v4i;

// ---- CDNA5 WMMA helpers (bf16 16x16x32, f32 accumulate) -------------------

static __device__ inline v16bf cat8(v8bf a, v8bf b) {
  return __builtin_shufflevector(a, b, 0,1,2,3,4,5,6,7,8,9,10,11,12,13,14,15);
}

// A-matrix fragment 16(M) x 32(K) bf16.  Lane(hl,hi) holds row M=hl,
// K = {8*hi+0..7} U {8*hi+16..23}  -> two contiguous 16B chunks.
static __device__ inline v16bf load_a_frag(const __bf16* base, int ld, int k0, int lane) {
  const int hl = lane & 15, hi = (lane >> 4) & 1;
  const __bf16* p = base + hl * ld + k0 + 8 * hi;
  return cat8(*(const v8bf*)p, *(const v8bf*)(p + 16));
}

// B-matrix fragment 32(K) x 16(N) bf16.  Lane(hl,hi) holds column N=hl,
// K = {8*hi+0..7} U {8*hi+16..23}  -> two contiguous 16B chunks.
static __device__ inline v16bf load_b_frag(const __bf16* base, int ld, int k0, int lane) {
  const int hl = lane & 15, hi = (lane >> 4) & 1;
  const __bf16* p = base + hl * ld + k0 + 8 * hi;
  return cat8(*(const v8bf*)p, *(const v8bf*)(p + 16));
}

static __device__ inline v8f wmma_bf16(v16bf a, v16bf b, v8f c) {
  return __builtin_amdgcn_wmma_f32_16x16x32_bf16(false, a, false, b, (short)0, c,
                                                 false, false);
}

// ---- Kernel 1: fused QKV projection GEMM ----------------------------------
// Y[m,o] = sum_c x[m,c]*w[o,c] + b[o];  M=8192, K=768, O=2304.
// 128x128 block tile, double-buffered LDS, 8 waves in a 4(M)x2(O) grid,
// each wave computes 32x64 (8 WMMA tiles) per 32-K step.
__global__ __launch_bounds__(256) void qkv_kernel(
    const float* __restrict__ x, const float* __restrict__ w,
    const float* __restrict__ bias,
    __bf16* __restrict__ qw, __bf16* __restrict__ kw, __bf16* __restrict__ vtw)
{
  __shared__ __bf16 Xs[2 * 128 * 32];   // 16 KB
  __shared__ __bf16 Ws[2 * 128 * 32];   // 16 KB
  const int tid = threadIdx.x;
  const int lane = tid & 31, wave = tid >> 5;
  const int wm = wave & 3, wn = wave >> 2;
  const int m0 = blockIdx.x * 128;
  const int o0 = blockIdx.y * 128;

  v8f acc[2][4];
#pragma unroll
  for (int i = 0; i < 2; ++i)
#pragma unroll
    for (int j = 0; j < 4; ++j) acc[i][j] = (v8f){0,0,0,0,0,0,0,0};

  float4 xr[4], wr[4];

  auto loadg = [&](int kt) {
#pragma unroll
    for (int i = 0; i < 4; ++i) {
      const int e = tid + i * 256;           // float4 index: 1024 per tile
      const int r = e >> 3, c = (e & 7) << 2;
      xr[i] = *(const float4*)&x[(size_t)(m0 + r) * CH + kt * 32 + c];
      wr[i] = *(const float4*)&w[(size_t)(o0 + r) * CH + kt * 32 + c];
    }
  };
  auto storeb = [&](int buf) {
#pragma unroll
    for (int i = 0; i < 4; ++i) {
      const int e = tid + i * 256;
      const int r = e >> 3, c = (e & 7) << 2;
      v4bf xv = { (__bf16)xr[i].x, (__bf16)xr[i].y, (__bf16)xr[i].z, (__bf16)xr[i].w };
      v4bf wv = { (__bf16)wr[i].x, (__bf16)wr[i].y, (__bf16)wr[i].z, (__bf16)wr[i].w };
      *(v4bf*)&Xs[buf * 4096 + r * 32 + c] = xv;
      *(v4bf*)&Ws[buf * 4096 + r * 32 + c] = wv;
    }
  };
  auto compute = [&](int buf) {
    const __bf16* xb = Xs + buf * 4096 + (wm * 32) * 32;
    const __bf16* wb = Ws + buf * 4096 + (wn * 64) * 32;
    v16bf a0 = load_a_frag(xb,       32, 0, lane);
    v16bf a1 = load_a_frag(xb + 512, 32, 0, lane);
#pragma unroll
    for (int tn = 0; tn < 4; ++tn) {
      v16bf b = load_b_frag(wb + tn * 512, 32, 0, lane);
      acc[0][tn] = wmma_bf16(a0, b, acc[0][tn]);
      acc[1][tn] = wmma_bf16(a1, b, acc[1][tn]);
    }
  };

  const int NT = CH / 32;                    // 24 K-steps
  loadg(0);
  storeb(0);
  __syncthreads();
  for (int kt = 0; kt < NT; ++kt) {
    const int cur = kt & 1;
    const bool more = (kt + 1 < NT);
    if (more) loadg(kt + 1);                 // overlap global latency with WMMA
    compute(cur);
    if (more) {
      __syncthreads();
      storeb(1 - cur);
      __syncthreads();
    }
  }

  // epilogue: scatter bf16 q[b,h,n,d], k[b,h,n,d], vT[b,h,d,n]
  const int hl = lane & 15, hi = (lane >> 4) & 1;
#pragma unroll
  for (int tm = 0; tm < 2; ++tm)
#pragma unroll
    for (int tn = 0; tn < 4; ++tn) {
      const int o = o0 + wn * 64 + tn * 16 + hl;
      const int sec = o / CH, c = o % CH;
      const int h = c >> 6, d = c & 63;
      const float bv = bias[o];
#pragma unroll
      for (int r = 0; r < 8; ++r) {
        const int m = m0 + wm * 32 + tm * 16 + r + 8 * hi;
        const int bidx = m >> 10, nr = m & 1023;
        const float val = acc[tm][tn][r] + bv;
        const size_t bh = (size_t)(bidx * HEADS + h);
        if (sec == 0)      qw [bh * SEQ * HD + (size_t)nr * HD + d] = (__bf16)val;
        else if (sec == 1) kw [bh * SEQ * HD + (size_t)nr * HD + d] = (__bf16)val;
        else               vtw[bh * HD * SEQ + (size_t)d * SEQ + nr] = (__bf16)val;
      }
    }
}

// ---- Kernel 2: attention core (S = QK^T, softmax, attn out, O = P V) ------
// One block per (b*H+h, 16-query-row tile). S kept in registers.
// Q tile fetched via the Tensor Data Mover (contiguous 2 KB block).
__global__ __launch_bounds__(256) void attn_kernel(
    const __bf16* __restrict__ qw, const __bf16* __restrict__ kw,
    const __bf16* __restrict__ vtw,
    float* __restrict__ attn_out, __bf16* __restrict__ o_ws)
{
  __shared__ __bf16 q_lds[16 * HD];        //  2 KB
  __shared__ __bf16 p_lds[16 * SEQ];       // 32 KB (unnormalized exp, bf16)
  __shared__ float  obuf[2 * 16 * HD];     //  8 KB (PV partials, 2 K-halves)
  __shared__ float  wred[8 * 16];          // cross-wave row reduction
  __shared__ float  rmax_lds[16];
  __shared__ float  rsumi_lds[16];

  const int tid = threadIdx.x, lane = tid & 31, wave = tid >> 5;
  const int hl = lane & 15, hi = (lane >> 4) & 1;
  const int bh = blockIdx.y;               // b*HEADS + h
  const int row0 = blockIdx.x * 16;
  const size_t qk_base = (size_t)bh * SEQ * HD;
  const float scale = 0.125f;              // 64^-0.5

  // ---- TDM: DMA the contiguous 16x64 bf16 Q tile into LDS (one wave issues).
  if (wave == 0) {
    const unsigned lds_off = (unsigned)(size_t)(&q_lds[0]);   // low 32b = LDS offset
    const unsigned long long ga =
        (unsigned long long)(size_t)(qw + qk_base + (size_t)row0 * HD);
    // D# group0: count=1 | lds_addr | global_addr[31:0] | addr[56:32], type=2
    v4u g0 = { 1u, lds_off, (unsigned)ga,
               (unsigned)((ga >> 32) & 0x1FFFFFFu) | 0x80000000u };
    // D# group1: data_size=2B; tensor_dim0=1024; tensor_dim1=1; tile_dim0=1024;
    //            tile_dim1=1; tensor_dim0_stride=1024 (1-D contiguous run)
    v8i g1 = { 0x10000, 1024 << 16, 1 << 16, 1024 << 16, 1, 1024, 0, 0 };
    v4i gz4 = { 0, 0, 0, 0 };
    v8i gz8 = { 0, 0, 0, 0, 0, 0, 0, 0 };
    __builtin_amdgcn_tensor_load_to_lds(g0, g1, gz4, gz4, gz8, 0);
    __builtin_amdgcn_s_wait_tensorcnt(0);
  }
  __syncthreads();

  // ---- S tile: each wave owns 128 key columns (8 WMMA tiles), K=HD=64.
  v8f accv[8];
  const int colbase = wave * 128;
  for (int ct = 0; ct < 8; ++ct) {
    v8f acc = {0,0,0,0,0,0,0,0};
    const __bf16* kb = kw + qk_base + (size_t)(colbase + ct * 16) * HD;
#pragma unroll
    for (int k0 = 0; k0 < HD; k0 += 32) {
      v16bf a = load_a_frag(q_lds, HD, k0, lane);
      v16bf b = load_b_frag(kb, HD, k0, lane);   // B column n = key row (contig d)
      acc = wmma_bf16(a, b, acc);
    }
#pragma unroll
    for (int r = 0; r < 8; ++r) acc[r] *= scale;
    accv[ct] = acc;
  }

  // ---- row max: lane partial -> butterfly over hl (16 lanes) -> LDS over waves
  float pmax[8];
#pragma unroll
  for (int r = 0; r < 8; ++r) {
    float mx = -3.4e38f;
#pragma unroll
    for (int ct = 0; ct < 8; ++ct) mx = fmaxf(mx, accv[ct][r]);
    for (int m = 1; m < 16; m <<= 1) mx = fmaxf(mx, __shfl_xor(mx, m));
    pmax[r] = mx;
  }
  if (hl == 0)
#pragma unroll
    for (int r = 0; r < 8; ++r) wred[wave * 16 + r + 8 * hi] = pmax[r];
  __syncthreads();
  if (tid < 16) {
    float mx = wred[tid];
#pragma unroll
    for (int w = 1; w < 8; ++w) mx = fmaxf(mx, wred[w * 16 + tid]);
    rmax_lds[tid] = mx;
  }
  __syncthreads();

  // ---- exp (unnormalized), stage bf16 probs for PV, reduce row sums
  float psum[8] = {0, 0, 0, 0, 0, 0, 0, 0};
  for (int ct = 0; ct < 8; ++ct)
#pragma unroll
    for (int r = 0; r < 8; ++r) {
      const int m = r + 8 * hi;
      const int col = colbase + ct * 16 + hl;
      const float pu = __expf(accv[ct][r] - rmax_lds[m]);
      accv[ct][r] = pu;
      p_lds[m * SEQ + col] = (__bf16)pu;
      psum[r] += pu;
    }
#pragma unroll
  for (int r = 0; r < 8; ++r)
    for (int m = 1; m < 16; m <<= 1) psum[r] += __shfl_xor(psum[r], m);
  if (hl == 0)
#pragma unroll
    for (int r = 0; r < 8; ++r) wred[wave * 16 + r + 8 * hi] = psum[r];
  __syncthreads();
  if (tid < 16) {
    float s = wred[tid];
#pragma unroll
    for (int w = 1; w < 8; ++w) s += wred[w * 16 + tid];
    rsumi_lds[tid] = 1.0f / s;
  }
  __syncthreads();

  // ---- write normalized attention probabilities (fp32, exact softmax)
  float* ab = attn_out + (size_t)bh * SEQ * SEQ + (size_t)row0 * SEQ;
  for (int ct = 0; ct < 8; ++ct)
#pragma unroll
    for (int r = 0; r < 8; ++r) {
      const int m = r + 8 * hi;
      ab[(size_t)m * SEQ + colbase + ct * 16 + hl] = accv[ct][r] * rsumi_lds[m];
    }

  // ---- O = P V : waves split K in halves of 512; each owns one 16x16 d-tile
  const int half = wave >> 2, dt = wave & 3;
  v8f acc = {0,0,0,0,0,0,0,0};
  const __bf16* vb = vtw + (size_t)bh * HD * SEQ + (size_t)(dt * 16) * SEQ;
  for (int kt = 0; kt < 16; ++kt) {
    const int k0 = half * 512 + kt * 32;
    v16bf a = load_a_frag(p_lds, SEQ, k0, lane);   // P rows 0..15
    v16bf b = load_b_frag(vb, SEQ, k0, lane);      // vT: contiguous along K
    acc = wmma_bf16(a, b, acc);
  }
#pragma unroll
  for (int r = 0; r < 8; ++r)
    obuf[(half * 16 + r + 8 * hi) * HD + dt * 16 + hl] = acc[r];
  __syncthreads();

  // ---- combine halves, apply 1/rowsum, store O bf16 as [B, N, C]
  const int b_ = bh / HEADS, h_ = bh % HEADS;
  for (int e = tid; e < 16 * HD; e += 256) {
    const int m = e >> 6, d = e & 63;
    const float val = (obuf[m * HD + d] + obuf[(16 + m) * HD + d]) * rsumi_lds[m];
    o_ws[((size_t)b_ * SEQ + row0 + m) * CH + h_ * HD + d] = (__bf16)val;
  }
}

// ---- Kernel 3: output projection GEMM -------------------------------------
// out[m,o] = sum_c O[m,c]*proj_w[o,c] + proj_b[o];  M=8192, K=768, O=768.
__global__ __launch_bounds__(256) void proj_kernel(
    const __bf16* __restrict__ oin, const float* __restrict__ w,
    const float* __restrict__ bias, float* __restrict__ out)
{
  __shared__ __bf16 Xs[2 * 128 * 32];
  __shared__ __bf16 Ws[2 * 128 * 32];
  const int tid = threadIdx.x;
  const int lane = tid & 31, wave = tid >> 5;
  const int wm = wave & 3, wn = wave >> 2;
  const int m0 = blockIdx.x * 128;
  const int o0 = blockIdx.y * 128;

  v8f acc[2][4];
#pragma unroll
  for (int i = 0; i < 2; ++i)
#pragma unroll
    for (int j = 0; j < 4; ++j) acc[i][j] = (v8f){0,0,0,0,0,0,0,0};

  v8bf ar[2];
  float4 wr[4];

  auto loadg = [&](int kt) {
#pragma unroll
    for (int i = 0; i < 2; ++i) {                // A is already bf16: 16B chunks
      const int e = tid + i * 256;               // 512 chunks of 8 elems
      const int r = e >> 2, c = (e & 3) << 3;
      ar[i] = *(const v8bf*)&oin[(size_t)(m0 + r) * CH + kt * 32 + c];
    }
#pragma unroll
    for (int i = 0; i < 4; ++i) {
      const int e = tid + i * 256;
      const int r = e >> 3, c = (e & 7) << 2;
      wr[i] = *(const float4*)&w[(size_t)(o0 + r) * CH + kt * 32 + c];
    }
  };
  auto storeb = [&](int buf) {
#pragma unroll
    for (int i = 0; i < 2; ++i) {
      const int e = tid + i * 256;
      const int r = e >> 2, c = (e & 3) << 3;
      *(v8bf*)&Xs[buf * 4096 + r * 32 + c] = ar[i];
    }
#pragma unroll
    for (int i = 0; i < 4; ++i) {
      const int e = tid + i * 256;
      const int r = e >> 3, c = (e & 7) << 2;
      v4bf wv = { (__bf16)wr[i].x, (__bf16)wr[i].y, (__bf16)wr[i].z, (__bf16)wr[i].w };
      *(v4bf*)&Ws[buf * 4096 + r * 32 + c] = wv;
    }
  };
  auto compute = [&](int buf) {
    const __bf16* xb = Xs + buf * 4096 + (wm * 32) * 32;
    const __bf16* wb = Ws + buf * 4096 + (wn * 64) * 32;
    v16bf a0 = load_a_frag(xb,       32, 0, lane);
    v16bf a1 = load_a_frag(xb + 512, 32, 0, lane);
#pragma unroll
    for (int tn = 0; tn < 4; ++tn) {
      v16bf b = load_b_frag(wb + tn * 512, 32, 0, lane);
      acc[0][tn] = wmma_bf16(a0, b, acc[0][tn]);
      acc[1][tn] = wmma_bf16(a1, b, acc[1][tn]);
    }
  };

  const int NT = CH / 32;
  loadg(0);
  storeb(0);
  __syncthreads();
  for (int kt = 0; kt < NT; ++kt) {
    const int cur = kt & 1;
    const bool more = (kt + 1 < NT);
    if (more) loadg(kt + 1);
    compute(cur);
    if (more) {
      __syncthreads();
      storeb(1 - cur);
      __syncthreads();
    }
  }

  const int hl = lane & 15, hi = (lane >> 4) & 1;
#pragma unroll
  for (int tm = 0; tm < 2; ++tm)
#pragma unroll
    for (int tn = 0; tn < 4; ++tn) {
      const int o = o0 + wn * 64 + tn * 16 + hl;
      const float bv = bias[o];
#pragma unroll
      for (int r = 0; r < 8; ++r) {
        const int m = m0 + wm * 32 + tm * 16 + r + 8 * hi;
        out[(size_t)m * CH + o] = acc[tm][tn][r] + bv;
      }
    }
}

// ---- host launch ----------------------------------------------------------
extern "C" void kernel_launch(void* const* d_in, const int* in_sizes, int n_in,
                              void* d_out, int out_size, void* d_ws, size_t ws_size,
                              hipStream_t stream) {
  const float* x      = (const float*)d_in[0];
  const float* qkv_w  = (const float*)d_in[1];
  const float* qkv_b  = (const float*)d_in[2];
  const float* proj_w = (const float*)d_in[3];
  const float* proj_b = (const float*)d_in[4];

  float* out  = (float*)d_out;                       // [B,N,C]
  float* attn = out + (size_t)BATCH * SEQ * CH;      // [B,H,N,N]

  const size_t per = (size_t)BATCH * HEADS * SEQ * HD;  // elements per q/k/vT
  __bf16* qw  = (__bf16*)d_ws;
  __bf16* kw  = qw  + per;
  __bf16* vtw = kw  + per;
  __bf16* ow  = vtw + per;                           // [B,N,C] bf16

  qkv_kernel<<<dim3((BATCH * SEQ) / 128, OC / 128), 256, 0, stream>>>(
      x, qkv_w, qkv_b, qw, kw, vtw);
  attn_kernel<<<dim3(SEQ / 16, BATCH * HEADS), 256, 0, stream>>>(
      qw, kw, vtw, attn, ow);
  proj_kernel<<<dim3((BATCH * SEQ) / 128, CH / 128), 256, 0, stream>>>(
      ow, proj_w, proj_b, out);
}
